// LSTMPredictor_68272800137459
// MI455X (gfx1250) — matrix operand order repaired and checked by reading, not compiled, not run on hardware
//
#include <hip/hip_runtime.h>
#include <hip/hip_bf16.h>

// ---------------------------------------------------------------------------
// LSTMPredictor on MI455X (gfx1250, wave32, WMMA):
//   persistent single-workgroup kernel. Weights pre-swizzled once into the
//   v_wmma_f32_16x16x32_f16 B-operand layout; cell-2 weights live in VGPRs,
//   cell-1 weights in LDS. Hidden state carried as f16 (A-operand row 0),
//   cell state in f32. Activations use hardware v_tanh_f32 when available.
//   Tile loops are split so the hot path has no exec-mask churn: tiles 0..11
//   run unconditionally, tile 12 is a wave0-only tail; gate stores are
//   unconditional into a stride-32 layout (upper lanes provably write 0).
// ---------------------------------------------------------------------------

typedef __attribute__((ext_vector_type(16))) _Float16 v16h;
typedef __attribute__((ext_vector_type(8)))  float    v8f;

#define TSEQ    8192
#define HID     51
#define GATE    204          // 4*H
#define NT      13           // ceil(204/16) output tiles
#define KT1     2            // K tiles for cell1 (K=51 padded to 64)
#define KT2     4            // K tiles for cell2 ([h1;h2] in 128-wide K space)
#define TILE_DW 256          // dwords per B tile (32 lanes * 8 dwords)
#define BLK     128          // 4 waves
#define NTPW    4            // max N-tiles per wave (3 common + 1 tail)
#define Z_DW    416          // 13 tiles * 32 lanes gate buffer

union HU { v16h v; unsigned u[8]; };

__device__ __forceinline__ unsigned pack2f16(float lo, float hi) {
  union { _Float16 h[2]; unsigned u; } p;
  p.h[0] = (_Float16)lo; p.h[1] = (_Float16)hi;
  return p.u;
}

// Branch-free activations. gfx1250 has a hardware transcendental TANH.
__device__ __forceinline__ float fast_tanh(float x) {
#if __has_builtin(__builtin_amdgcn_tanhf)
  return __builtin_amdgcn_tanhf(x);
#else
  x = fminf(fmaxf(x, -20.f), 20.f);
  float e = __expf(-2.0f * x);
  return (1.0f - e) * __builtin_amdgcn_rcpf(1.0f + e);
#endif
}
__device__ __forceinline__ float fast_sig(float x) {
  return 0.5f * fast_tanh(0.5f * x) + 0.5f;
}

// gate g lives at z[(g/16)*32 + g%16] (stride-32 tile rows)
__device__ __forceinline__ int zidx(int g) { return ((g >> 4) << 5) | (g & 15); }

// A operand: h vector in matrix row M=0 of a 16x32 f16 tile.
// 16-bit A layout: lane0 holds K=[0..7] (v0..3) and K=[16..23] (v4..7);
// lane16 holds K=[8..15] and K=[24..31]; all other lanes (rows M>0) zero.
__device__ __forceinline__ v16h build_a(const unsigned* hdw, int kt, int lane) {
  HU a;
  const int  base = kt * 16 + ((lane >> 4) << 2);
  const bool row0 = (lane & 15) == 0;
#pragma unroll
  for (int i = 0; i < 4; ++i) {
    a.u[i]     = row0 ? hdw[base + i]     : 0u;
    a.u[4 + i] = row0 ? hdw[base + 8 + i] : 0u;
  }
  return a.v;
}

// B tiles stored pre-swizzled: [tile][lane][8 dwords] -> two ds_load_b128.
__device__ __forceinline__ v16h load_b(const unsigned* tbase, int lane) {
  HU b;
  const unsigned* p = tbase + lane * 8;
#pragma unroll
  for (int i = 0; i < 8; ++i) b.u[i] = p[i];
  return b.v;
}

extern __shared__ __align__(16) unsigned smem_u[];

__global__ void
__launch_bounds__(BLK)
lstm_persistent_kernel(const float* __restrict__ x,
                       const float* __restrict__ Wih1, const float* __restrict__ Whh1,
                       const float* __restrict__ bih1, const float* __restrict__ bhh1,
                       const float* __restrict__ Wih2, const float* __restrict__ Whh2,
                       const float* __restrict__ bih2, const float* __restrict__ bhh2,
                       const float* __restrict__ Wlin, const float* __restrict__ blin,
                       float* __restrict__ out) {
  const int tid  = threadIdx.x;
  const int lane = tid & 31;
  const int wid  = tid >> 5;

  // ---- LDS partition -------------------------------------------------------
  unsigned* b1t = smem_u;                          // NT*KT1*256 dw  (26 KB)
  unsigned* b2t = b1t + NT * KT1 * TILE_DW;        // NT*KT2*256 dw  (52 KB, staging)
  float* b1f  = (float*)(b2t + NT * KT2 * TILE_DW);// [224] bih1+bhh1 (padded)
  float* wcol = b1f + 224;                         // [224] Wih1 column
  float* b2f  = wcol + 224;                        // [224] bih2+bhh2
  float* z1   = b2f + 224;                         // [416] cell1 gates (stride-32)
  float* z2   = z1 + Z_DW;                         // [416] cell2 gates (stride-32)
  float* c1s  = z2 + Z_DW;                         // [64] cell1 c state (f32)
  float* c2s  = c1s + 64;                          // [64] cell2 c state (f32)
  float* h2f  = c2s + 64;                          // [64] h2 in f32 (out dot)
  float* wl   = h2f + 64;                          // [64] Wlin
  unsigned* hdw = (unsigned*)(wl + 64);            // [64] dw = 128 f16: h1|h2

  // ---- One-time weight pre-pack into WMMA B-operand layout ----------------
  // B[K][N] register map: VGPR r, lane l -> K = kbase + (l>=16?8:0) + 2(r&3) + 16(r>>2) (+1),
  //                                         N = ntile*16 + (l&15)
  for (int idx = tid; idx < NT * KT1 * TILE_DW; idx += BLK) {
    int tt = idx >> 8, s = idx & 255;
    int nt = tt >> 1, kt = tt & 1;
    int l = s >> 3, r = s & 7;
    int K = kt * 32 + ((l >> 4) << 3) + 2 * (r & 3) + ((r >> 2) << 4);
    int N = nt * 16 + (l & 15);
    float lo = (K     < HID && N < GATE) ? Whh1[N * HID + K]     : 0.f;
    float hi = (K + 1 < HID && N < GATE) ? Whh1[N * HID + K + 1] : 0.f;
    b1t[idx] = pack2f16(lo, hi);
  }
  // cell2: fused [Wih2 | Whh2]; K<64 -> h1 slot, K>=64 -> h2 slot
  for (int idx = tid; idx < NT * KT2 * TILE_DW; idx += BLK) {
    int tt = idx >> 8, s = idx & 255;
    int nt = tt >> 2, kt = tt & 3;
    int l = s >> 3, r = s & 7;
    int K = kt * 32 + ((l >> 4) << 3) + 2 * (r & 3) + ((r >> 2) << 4);
    int N = nt * 16 + (l & 15);
    float lo = 0.f, hi = 0.f;
    if (N < GATE) {
      int k0 = K, k1 = K + 1;
      lo = (k0 < 64) ? ((k0 < HID) ? Wih2[N * HID + k0] : 0.f)
                     : ((k0 - 64 < HID) ? Whh2[N * HID + (k0 - 64)] : 0.f);
      hi = (k1 < 64) ? ((k1 < HID) ? Wih2[N * HID + k1] : 0.f)
                     : ((k1 - 64 < HID) ? Whh2[N * HID + (k1 - 64)] : 0.f);
    }
    b2t[idx] = pack2f16(lo, hi);
  }
  for (int i = tid; i < 224; i += BLK) {
    b1f[i]  = (i < GATE) ? (bih1[i] + bhh1[i]) : 0.f;
    wcol[i] = (i < GATE) ? Wih1[i] : 0.f;   // Wih1 is (204,1)
    b2f[i]  = (i < GATE) ? (bih2[i] + bhh2[i]) : 0.f;
  }
  for (int i = tid; i < Z_DW; i += BLK) { z1[i] = 0.f; z2[i] = 0.f; }
  for (int i = tid; i < 64; i += BLK) {
    c1s[i] = 0.f; c2s[i] = 0.f; h2f[i] = 0.f;
    wl[i] = (i < HID) ? Wlin[i] : 0.f;
    hdw[i] = 0u;                            // h1 = h2 = 0 (f16, padded)
  }
  const float blin0 = blin[0];
  __syncthreads();

  // ---- Register-cache loop-invariant operands -----------------------------
  // Per-wave bias values (row 0 of the C operand) and the *entire* cell-2
  // weight set (16 x v16h = 128 VGPRs/wave; occupancy is irrelevant here).
  float biasx1[NTPW], wcolr[NTPW], biasr2[NTPW];
  v16h  B2r[NTPW][KT2];
#pragma unroll
  for (int i = 0; i < NTPW; ++i) {
    const int ntc = (wid + i * 4 < NT) ? (wid + i * 4) : 0;
    const int n   = ntc * 16 + (lane & 15);
    biasx1[i] = b1f[n];
    wcolr[i]  = wcol[n];
    biasr2[i] = b2f[n];
#pragma unroll
    for (int kt = 0; kt < KT2; ++kt)
      B2r[i][kt] = load_b(b2t + (ntc * KT2 + kt) * TILE_DW, lane);
  }

  // ---- Sequential scan over 8192 timesteps --------------------------------
  for (int t = 0; t < TSEQ; ++t) {
    const float xt = x[t];
    __builtin_prefetch(&x[(t + 64) & (TSEQ - 1)], 0, 0);

    // cell1: z1 = Whh1 @ h1 + (bih1+bhh1 + Wih1*xt); h1 in A row 0, B in LDS
    {
      v16h a0 = build_a(hdw, 0, lane);
      v16h a1 = build_a(hdw, 1, lane);
#pragma unroll
      for (int i = 0; i < 3; ++i) {                     // nt = wid+4i <= 11: no guard
        const int nt = wid + i * 4;
        v8f c = {};
        c[0] = (lane < 16) ? (biasx1[i] + wcolr[i] * xt) : 0.f;
        v16h bb0 = load_b(b1t + (nt * KT1 + 0) * TILE_DW, lane);
        c = __builtin_amdgcn_wmma_f32_16x16x32_f16(false, a0, false, bb0,
                                                   (short)0, c, false, false);
        v16h bb1 = load_b(b1t + (nt * KT1 + 1) * TILE_DW, lane);
        c = __builtin_amdgcn_wmma_f32_16x16x32_f16(false, a1, false, bb1,
                                                   (short)0, c, false, false);
        z1[nt * 32 + lane] = c[0];      // lanes>=16 store provable zeros
      }
      if (wid == 0) {                                    // tail tile nt = 12
        v8f c = {};
        c[0] = (lane < 16) ? (biasx1[3] + wcolr[3] * xt) : 0.f;
        v16h bb0 = load_b(b1t + (12 * KT1 + 0) * TILE_DW, lane);
        c = __builtin_amdgcn_wmma_f32_16x16x32_f16(false, a0, false, bb0,
                                                   (short)0, c, false, false);
        v16h bb1 = load_b(b1t + (12 * KT1 + 1) * TILE_DW, lane);
        c = __builtin_amdgcn_wmma_f32_16x16x32_f16(false, a1, false, bb1,
                                                   (short)0, c, false, false);
        z1[12 * 32 + lane] = c[0];
      }
    }
    __syncthreads();

    // cell1 activations (PyTorch gate order i,f,g,o); c kept in f32
    if (tid < HID) {
      const int j = tid;
      float zi = z1[zidx(j)],           zf = z1[zidx(HID + j)];
      float zg = z1[zidx(2 * HID + j)], zo = z1[zidx(3 * HID + j)];
      float cn = fast_sig(zf) * c1s[j] + fast_sig(zi) * fast_tanh(zg);
      float hn = fast_sig(zo) * fast_tanh(cn);
      c1s[j] = cn;
      ((_Float16*)hdw)[j] = (_Float16)hn;      // h1 -> A-operand f16 slot
    }
    __syncthreads();

    // cell2: z2 = [Wih2|Whh2] @ [h1;h2] + b2; B fully register-resident
    {
      v16h A[KT2];
#pragma unroll
      for (int kt = 0; kt < KT2; ++kt) A[kt] = build_a(hdw, kt, lane);
#pragma unroll
      for (int i = 0; i < 3; ++i) {                     // nt = wid+4i <= 11: no guard
        const int nt = wid + i * 4;
        v8f c = {};
        c[0] = (lane < 16) ? biasr2[i] : 0.f;
#pragma unroll
        for (int kt = 0; kt < KT2; ++kt)
          c = __builtin_amdgcn_wmma_f32_16x16x32_f16(false, A[kt], false,
                                                     B2r[i][kt], (short)0, c,
                                                     false, false);
        z2[nt * 32 + lane] = c[0];
      }
      if (wid == 0) {                                    // tail tile nt = 12
        v8f c = {};
        c[0] = (lane < 16) ? biasr2[3] : 0.f;
#pragma unroll
        for (int kt = 0; kt < KT2; ++kt)
          c = __builtin_amdgcn_wmma_f32_16x16x32_f16(false, A[kt], false,
                                                     B2r[3][kt], (short)0, c,
                                                     false, false);
        z2[12 * 32 + lane] = c[0];
      }
    }
    __syncthreads();

    // cell2 activations; h2 kept f32 for the output dot + f16 for next step
    if (tid < HID) {
      const int j = tid;
      float zi = z2[zidx(j)],           zf = z2[zidx(HID + j)];
      float zg = z2[zidx(2 * HID + j)], zo = z2[zidx(3 * HID + j)];
      float cn = fast_sig(zf) * c2s[j] + fast_sig(zi) * fast_tanh(zg);
      float hn = fast_sig(zo) * fast_tanh(cn);
      c2s[j] = cn;
      ((_Float16*)hdw)[64 + j] = (_Float16)hn;
      h2f[j] = hn;
    }
    __syncthreads();

    // output: out[t] = Wlin . h2 + blin  (wave0; zero-padded -> branchless)
    if (tid < 32) {
      float p = h2f[lane] * wl[lane] + h2f[lane + 32] * wl[lane + 32];
#pragma unroll
      for (int off = 16; off; off >>= 1) p += __shfl_xor(p, off, 32);
      if (lane == 0) out[t] = p + blin0;
    }
    // next iteration's reads of hdw/z are fenced by the barriers above
  }
}

extern "C" void kernel_launch(void* const* d_in, const int* in_sizes, int n_in,
                              void* d_out, int out_size, void* d_ws, size_t ws_size,
                              hipStream_t stream) {
  (void)in_sizes; (void)n_in; (void)d_ws; (void)ws_size; (void)out_size;
  const float* x    = (const float*)d_in[0];
  const float* Wih1 = (const float*)d_in[1];
  const float* Whh1 = (const float*)d_in[2];
  const float* bih1 = (const float*)d_in[3];
  const float* bhh1 = (const float*)d_in[4];
  const float* Wih2 = (const float*)d_in[5];
  const float* Whh2 = (const float*)d_in[6];
  const float* bih2 = (const float*)d_in[7];
  const float* bhh2 = (const float*)d_in[8];
  const float* Wlin = (const float*)d_in[9];
  const float* blin = (const float*)d_in[10];
  float* out = (float*)d_out;

  constexpr int LDS_DW = NT * KT1 * TILE_DW + NT * KT2 * TILE_DW  // B tiles
                       + 224 * 3                                  // b1f,wcol,b2f
                       + Z_DW * 2                                 // z1,z2
                       + 64 * 4                                   // c1s,c2s,h2f,wl
                       + 64;                                      // hdw
  constexpr size_t SMEM_BYTES = (size_t)LDS_DW * 4;               // ~87 KB of 320 KB

  lstm_persistent_kernel<<<dim3(1), dim3(BLK), SMEM_BYTES, stream>>>(
      x, Wih1, Whh1, bih1, bhh1, Wih2, Whh2, bih2, bhh2, Wlin, blin, out);
}
